// PeriodicConvolution_1451698946408
// MI455X (gfx1250) — compile-verified, matrix operand order checked by reading, not run on hardware
//
#include <hip/hip_runtime.h>

typedef __bf16 bf16_t;
typedef __attribute__((ext_vector_type(16))) __bf16 v16bf;
typedef __attribute__((ext_vector_type(8)))  float  v8f;

#define PN    512
#define CIN   32
#define COUTC 32
#define HID   128
#define NB_N  64          // (COUT*CIN)/16 column blocks
#define KSTEPS 4          // 128 / 32
#define W2FRAG_ELEMS (NB_N * KSTEPS * 32 * 16)   // 131072 bf16 = 256 KB

// ---------------------------------------------------------------------------
// Prep: swizzle W2 [128][1024] f32 -> bf16 WMMA *A*-fragments of W2^T.
//   A row  M = (co,ci) column index = nb*16 + (lane&15)
//   A col  K = hid = ks*32 + ((lane>>4)*8) + (j<8 ? j : j+8)
// ---------------------------------------------------------------------------
__global__ void prep_w2_kernel(const float* __restrict__ W2, bf16_t* __restrict__ w2f) {
    int idx = blockIdx.x * blockDim.x + threadIdx.x;
    if (idx >= W2FRAG_ELEMS) return;
    int j    = idx & 15;
    int lane = (idx >> 4) & 31;
    int ks   = (idx >> 9) & 3;
    int nb   = idx >> 11;
    int n = nb * 16 + (lane & 15);
    int k = ks * 32 + ((lane >> 4) << 3) + (j < 8 ? j : j + 8);
    w2f[idx] = (bf16_t)W2[k * (COUTC * CIN) + n];
}

// ---------------------------------------------------------------------------
// Fused kernel, transposed mapping:
//   D = W2^T-chunk (A, LDS) @ h^T (B, registers)
//   D rows = 16 (co,ci) columns (VGPR slots), D cols = 16 edges (lanes)
//   msg[e,co]: v8f accumulators aligned with D (packed FMAs, no copies),
//   horizontal 8-sum + one xor-16 combine, one full-wave atomic per co.
// ---------------------------------------------------------------------------
__launch_bounds__(256)
__global__ void conv_kernel(const float* __restrict__ features,
                            const float* __restrict__ edge_vec,
                            const float* __restrict__ W1,
                            const float* __restrict__ b1,
                            const bf16_t* __restrict__ w2f,
                            const int* __restrict__ ebatch,
                            const int* __restrict__ ea,
                            const int* __restrict__ ebn,
                            const int* __restrict__ n_norm,
                            float* __restrict__ out, int nEdges)
{
    __shared__ bf16_t w2s[W2FRAG_ELEMS];   // 256 KB: whole W2 in fragment order

    const int tid = threadIdx.x;
    {   // coalesced bulk copy global -> LDS (16B per thread-iter)
        const uint4* src = (const uint4*)w2f;
        uint4*       dst = (uint4*)w2s;
        for (int i = tid; i < W2FRAG_ELEMS / 8; i += 256) dst[i] = src[i];
    }
    __syncthreads();

    const int lane    = tid & 31;
    const int halfsel = (lane >> 4) & 1;
    const int l15     = lane & 15;
    const int waveId  = blockIdx.x * (blockDim.x >> 5) + (tid >> 5);
    const int nWaves  = gridDim.x * (blockDim.x >> 5);
    const float scale = rsqrtf((float)n_norm[0]);
    const int nST     = (nEdges + 31) >> 5;

    for (int st = waveId; st < nST; st += nWaves) {   // wave-uniform: EXEC full
        const int eb0 = st << 5;

        // ---- per-lane edge for both tiles: e = eb0 + t*16 + (lane&15) ----
        float xv[2][4];
        #pragma unroll
        for (int t = 0; t < 2; ++t) {
            int e  = eb0 + t * 16 + l15;
            int ec = e < nEdges ? e : 0;
            float vx = edge_vec[ec * 3 + 0];
            float vy = edge_vec[ec * 3 + 1];
            float vz = edge_vec[ec * 3 + 2];
            float r  = sqrtf(vx * vx + vy * vy + vz * vz);
            bool zero = (r < 1e-10f) || (e >= nEdges);
            xv[t][0] = zero ? 0.f : vx;
            xv[t][1] = zero ? 0.f : vy;
            xv[t][2] = zero ? 0.f : vz;
            xv[t][3] = zero ? 0.f : r;
        }

        // ---- build h^T B-fragments: h = relu(x@W1+b1), WMMA 16-bit layout --
        v16bf hfrag[2][KSTEPS];
        #pragma unroll
        for (int ks = 0; ks < KSTEPS; ++ks) {
            const int kb = ks * 32 + (halfsel << 3);
            #pragma unroll
            for (int j = 0; j < 16; ++j) {
                int hid = kb + (j < 8 ? j : j + 8);
                float w0  = W1[0 * HID + hid];
                float w1v = W1[1 * HID + hid];
                float w2v = W1[2 * HID + hid];
                float w3v = W1[3 * HID + hid];
                float bb  = b1[hid];
                #pragma unroll
                for (int t = 0; t < 2; ++t) {
                    float hv = fmaf(xv[t][0], w0,
                               fmaf(xv[t][1], w1v,
                               fmaf(xv[t][2], w2v,
                               fmaf(xv[t][3], w3v, bb))));
                    hfrag[t][ks][j] = (bf16_t)fmaxf(hv, 0.f);
                }
            }
        }

        // ---- f_b (scaled) as 32B-aligned v8f + scatter segments ------------
        // lane needs ci = p*16 + 8*halfsel + i  (i=0..7) -> one v8f per (t,p)
        v8f fbv[2][2];
        int segb[2];
        #pragma unroll
        for (int t = 0; t < 2; ++t) {
            int e  = eb0 + t * 16 + l15;
            int ok = e < nEdges;
            int ec = ok ? e : 0;
            int bch   = ebatch[ec];
            int fbrow = (bch * PN + ebn[ec]) * CIN + (halfsel << 3);
            #pragma unroll
            for (int p = 0; p < 2; ++p)
                fbv[t][p] = (*(const v8f*)&features[fbrow + p * 16]) * scale;
            segb[t] = ok ? (bch * PN + ea[ec]) * COUTC : -1;
        }

        // ---- main fused GEMM over output channels -------------------------
        #pragma unroll 1
        for (int co = 0; co < COUTC; ++co) {
            v8f macc0 = {0.f, 0.f, 0.f, 0.f, 0.f, 0.f, 0.f, 0.f};
            v8f macc1 = {0.f, 0.f, 0.f, 0.f, 0.f, 0.f, 0.f, 0.f};
            #pragma unroll
            for (int p = 0; p < 2; ++p) {
                const int nb = co * 2 + p;
                v8f d0 = {0.f, 0.f, 0.f, 0.f, 0.f, 0.f, 0.f, 0.f};
                v8f d1 = {0.f, 0.f, 0.f, 0.f, 0.f, 0.f, 0.f, 0.f};
                #pragma unroll
                for (int ks = 0; ks < KSTEPS; ++ks) {
                    const v16bf afrag =
                        *(const v16bf*)&w2s[((nb * KSTEPS + ks) * 32 + lane) * 16];
                    d0 = __builtin_amdgcn_wmma_f32_16x16x32_bf16(
                             false, afrag, false, hfrag[0][ks], (short)0, d0, false, false);
                    d1 = __builtin_amdgcn_wmma_f32_16x16x32_bf16(
                             false, afrag, false, hfrag[1][ks], (short)0, d1, false, false);
                }
                // element-aligned packed FMAs: D slot i <-> ci = p*16+8*half+i
                macc0 += d0 * fbv[0][p];
                macc1 += d1 * fbv[1][p];
            }
            // horizontal sum of the 8 in-lane ci terms (pure VALU, no WMMA dep)
            float m0 = ((macc0[0] + macc0[1]) + (macc0[2] + macc0[3]))
                     + ((macc0[4] + macc0[5]) + (macc0[6] + macc0[7]));
            float m1 = ((macc1[0] + macc1[1]) + (macc1[2] + macc1[3]))
                     + ((macc1[4] + macc1[5]) + (macc1[6] + macc1[7]));
            // combine the two ci-halves (lane l <-> l^16), then scatter:
            // lanes 0-15 carry tile 0 edges, lanes 16-31 carry tile 1 edges.
            m0 += __shfl_xor(m0, 16, 32);
            m1 += __shfl_xor(m1, 16, 32);
            const int   sb = halfsel ? segb[1] : segb[0];
            const float mv = halfsel ? m1 : m0;
            if (sb >= 0)
                atomicAdd(&out[sb + co], mv);
        }
    }
}

// ---------------------------------------------------------------------------
extern "C" void kernel_launch(void* const* d_in, const int* in_sizes, int n_in,
                              void* d_out, int out_size, void* d_ws, size_t ws_size,
                              hipStream_t stream) {
    const float* features = (const float*)d_in[0];
    const float* edge_vec = (const float*)d_in[1];
    const float* W1       = (const float*)d_in[2];
    const float* b1       = (const float*)d_in[3];
    const float* W2       = (const float*)d_in[4];
    const int*   ebatch   = (const int*)d_in[5];
    const int*   ea       = (const int*)d_in[6];
    const int*   ebn      = (const int*)d_in[7];
    const int*   n_norm   = (const int*)d_in[8];
    float*       out      = (float*)d_out;
    const int    nE       = in_sizes[5];

    bf16_t* w2f = (bf16_t*)d_ws;   // 256 KB fragment buffer

    hipMemsetAsync(d_out, 0, (size_t)out_size * sizeof(float), stream);

    prep_w2_kernel<<<dim3((W2FRAG_ELEMS + 255) / 256), dim3(256), 0, stream>>>(W2, w2f);

    conv_kernel<<<dim3(64), dim3(256), 0, stream>>>(
        features, edge_vec, W1, b1, w2f, ebatch, ea, ebn, n_norm, out, nE);
}
